// FinancialNetworkModule_55808805044793
// MI455X (gfx1250) — matrix operand order repaired and checked by reading, not compiled, not run on hardware
//
#include <hip/hip_runtime.h>
#include <hip/hip_bf16.h>

typedef __attribute__((ext_vector_type(16))) __bf16 bf16x16;
typedef __attribute__((ext_vector_type(8)))  float  floatx8;

static constexpr int NN = 10000;    // nodes
static constexpr int NE = 160000;   // edges
static constexpr int NG = 16;       // graphs
static constexpr int HD = 256;      // hidden
static constexpr int OUTF = 512;

// ---------------- helpers ----------------
__device__ __forceinline__ unsigned short f2bf(float f) {
  union { float f; unsigned int u; } v; v.f = f;
  unsigned int u = v.u;
  u += 0x7FFFu + ((u >> 16) & 1u);           // round-to-nearest-even
  return (unsigned short)(u >> 16);
}

union FragU { bf16x16 v; uint4 q[2]; };

// A fragment (16x32 bf16, MxK): lanes 0-15 row=lane, K = kc..kc+15;
// lanes 16-31 same rows, K = kc+16..kc+31.  Two 16B loads.
__device__ __forceinline__ bf16x16 load_a(const unsigned short* rowp, int kc, int half) {
  FragU f;
  int k0 = kc + (half << 4);
  f.q[0] = *(const uint4*)(rowp + k0);
  f.q[1] = *(const uint4*)(rowp + k0 + 8);
  return f.v;
}
// B fragment (32x16 bf16, KxN) from N-major weights: lanes 0-15 col=lane hold K=kc..kc+15,
// lanes 16-31 hold K=kc+16..kc+31.  Two 16B loads.
__device__ __forceinline__ bf16x16 load_b(const unsigned short* colp, int kc, int half) {
  FragU f;
  int k0 = kc + (half << 4);
  f.q[0] = *(const uint4*)(colp + k0);
  f.q[1] = *(const uint4*)(colp + k0 + 8);
  return f.v;
}
__device__ __forceinline__ floatx8 wmma_bf(bf16x16 a, bf16x16 b, floatx8 c) {
  return __builtin_amdgcn_wmma_f32_16x16x32_bf16(false, a, false, b, (short)0, c, false, false);
}

// ---------------- prep kernels ----------------
__global__ void k_f2bf(const float* __restrict__ s, unsigned short* __restrict__ d, int n) {
  int i = blockIdx.x * 256 + threadIdx.x;
  if (i < n) d[i] = f2bf(s[i]);
}
// src [K,N] f32 -> dst [N,K] bf16
__global__ void k_tconv(const float* __restrict__ s, unsigned short* __restrict__ d, int K, int N) {
  int i = blockIdx.x * 256 + threadIdx.x;
  if (i < K * N) { int k = i / N, n = i % N; d[(size_t)n * K + k] = f2bf(s[i]); }
}
__global__ void k_zero(float* p, int n) {
  int i = blockIdx.x * 256 + threadIdx.x;
  if (i < n) p[i] = 0.0f;
}
__global__ void k_deg(const int* __restrict__ dst, float* deg, int ne) {
  int e = blockIdx.x * 256 + threadIdx.x;
  if (e < ne) atomicAdd(&deg[dst[e]], 1.0f);
}

// ---------------- encoder: GEMM(K)->bias->LN->ReLU, N=256 ----------------
__global__ __launch_bounds__(256) void k_encode(
    const unsigned short* __restrict__ A, const unsigned short* __restrict__ Bt,
    const float* __restrict__ bias, const float* __restrict__ gamma,
    const float* __restrict__ beta, unsigned short* __restrict__ outb,
    float* __restrict__ outf, int K) {
  __shared__ float sm[16][HD];
  __shared__ float red[2][16][16];
  __shared__ float mn[16], iv[16];
  int wave = threadIdx.x >> 5, lane = threadIdx.x & 31;
  int half = lane >> 4, l15 = lane & 15;
  int row0 = blockIdx.x * 16;
  const unsigned short* arow = A + (size_t)(row0 + l15) * K;
  const unsigned short* bcol0 = Bt + (size_t)(wave * 32 + l15) * K;
  const unsigned short* bcol1 = bcol0 + (size_t)16 * K;
  floatx8 acc0 = {}, acc1 = {};
  for (int kc = 0; kc < K; kc += 32) {
    bf16x16 a = load_a(arow, kc, half);
    acc0 = wmma_bf(a, load_b(bcol0, kc, half), acc0);
    acc1 = wmma_bf(a, load_b(bcol1, kc, half), acc1);
  }
#pragma unroll
  for (int t = 0; t < 2; ++t) {
    int n = (wave * 2 + t) * 16 + l15;
    float bn = bias[n];
    floatx8 c = t ? acc1 : acc0;
#pragma unroll
    for (int r = 0; r < 8; ++r) sm[r + half * 8][n] = c[r] + bn;
  }
  __syncthreads();
  int rr = threadIdx.x >> 4, sub = threadIdx.x & 15;
  float s = 0.f, s2 = 0.f;
#pragma unroll
  for (int j = 0; j < 16; ++j) { float v = sm[rr][sub * 16 + j]; s += v; s2 += v * v; }
  red[0][rr][sub] = s; red[1][rr][sub] = s2;
  __syncthreads();
  if (sub == 0) {
    float ts = 0.f, ts2 = 0.f;
    for (int j = 0; j < 16; ++j) { ts += red[0][rr][j]; ts2 += red[1][rr][j]; }
    float m = ts / 256.f, var = ts2 / 256.f - m * m;
    mn[rr] = m; iv[rr] = rsqrtf(var + 1e-5f);
  }
  __syncthreads();
  float m = mn[rr], inv = iv[rr];
  size_t ob = (size_t)(row0 + rr) * HD;
#pragma unroll
  for (int j = 0; j < 16; ++j) {
    int n = sub * 16 + j;
    float v = (sm[rr][n] - m) * inv * gamma[n] + beta[n];
    v = fmaxf(v, 0.f);
    outb[ob + n] = f2bf(v);
    if (outf) outf[ob + n] = v;
  }
}

// ------- fused edge MLP: gather-concat GEMM1(768)->ReLU->LDS->GEMM2(256)->atomic agg -------
__global__ __launch_bounds__(256) void k_message(
    const unsigned short* __restrict__ xb, const unsigned short* __restrict__ eb,
    const int* __restrict__ src, const int* __restrict__ dst,
    const unsigned short* __restrict__ W1t, const float* __restrict__ b1,
    const unsigned short* __restrict__ W2t, const float* __restrict__ b2,
    float* __restrict__ agg) {
  __shared__ unsigned short hid[16][HD];   // 16 edges x 256 hidden, bf16
  int wave = threadIdx.x >> 5, lane = threadIdx.x & 31;
  int half = lane >> 4, l15 = lane & 15;
  int e0 = blockIdx.x * 16;
  int ea = e0 + l15;
  const unsigned short* seg0 = xb + (size_t)src[ea] * HD;
  const unsigned short* seg1 = xb + (size_t)dst[ea] * HD;
  const unsigned short* seg2 = eb + (size_t)ea * HD;
  {
    const unsigned short* bcol0 = W1t + (size_t)(wave * 32 + l15) * 768;
    const unsigned short* bcol1 = bcol0 + (size_t)16 * 768;
    floatx8 acc0 = {}, acc1 = {};
    for (int kc = 0; kc < 768; kc += 32) {
      const unsigned short* arow = (kc < 256) ? seg0 : (kc < 512) ? seg1 : seg2;
      bf16x16 a = load_a(arow, kc & 255, half);
      acc0 = wmma_bf(a, load_b(bcol0, kc, half), acc0);
      acc1 = wmma_bf(a, load_b(bcol1, kc, half), acc1);
    }
#pragma unroll
    for (int t = 0; t < 2; ++t) {
      int n = (wave * 2 + t) * 16 + l15;
      float bn = b1[n];
      floatx8 c = t ? acc1 : acc0;
#pragma unroll
      for (int r = 0; r < 8; ++r) hid[r + half * 8][n] = f2bf(fmaxf(c[r] + bn, 0.f));
    }
  }
  __syncthreads();
  {
    const unsigned short* bcol0 = W2t + (size_t)(wave * 32 + l15) * HD;
    const unsigned short* bcol1 = bcol0 + (size_t)16 * HD;
    floatx8 acc0 = {}, acc1 = {};
    for (int kc = 0; kc < HD; kc += 32) {
      bf16x16 a = load_a(&hid[l15][0], kc, half);
      acc0 = wmma_bf(a, load_b(bcol0, kc, half), acc0);
      acc1 = wmma_bf(a, load_b(bcol1, kc, half), acc1);
    }
#pragma unroll
    for (int t = 0; t < 2; ++t) {
      int n = (wave * 2 + t) * 16 + l15;
      float bn = b2[n];
      floatx8 c = t ? acc1 : acc0;
#pragma unroll
      for (int r = 0; r < 8; ++r) {
        int e = e0 + r + half * 8;
        atomicAdd(&agg[(size_t)dst[e] * HD + n], c[r] + bn);
      }
    }
  }
}

__global__ void k_aggfix(const float* __restrict__ agg, const float* __restrict__ deg,
                         unsigned short* __restrict__ aggb) {
  int i = blockIdx.x, h = threadIdx.x;
  aggb[(size_t)i * HD + h] = f2bf(agg[(size_t)i * HD + h] / fmaxf(deg[i], 1.0f));
}

// ------- generic GEMM + bias -> f32 (one wave, 16x64 tile: A-frag reused over 4 N-tiles) -------
__global__ __launch_bounds__(32) void k_gemm(
    const unsigned short* __restrict__ A, const unsigned short* __restrict__ Bt,
    const float* __restrict__ bias, float* __restrict__ out, int K, int N) {
  int lane = threadIdx.x & 31, half = lane >> 4, l15 = lane & 15;
  const unsigned short* arow = A + (size_t)(blockIdx.x * 16 + l15) * K;
  const unsigned short* bc = Bt + (size_t)(blockIdx.y * 64 + l15) * K;
  floatx8 acc[4] = {{}, {}, {}, {}};
  for (int kc = 0; kc < K; kc += 32) {
    bf16x16 a = load_a(arow, kc, half);
#pragma unroll
    for (int j = 0; j < 4; ++j)
      acc[j] = wmma_bf(a, load_b(bc + (size_t)j * 16 * K, kc, half), acc[j]);
  }
#pragma unroll
  for (int j = 0; j < 4; ++j) {
    int n = blockIdx.y * 64 + j * 16 + l15;
    float bn = bias[n];
#pragma unroll
    for (int r = 0; r < 8; ++r)
      out[(size_t)(blockIdx.x * 16 + r + half * 8) * N + n] = acc[j][r] + bn;
  }
}

// ---------------- GRU gates + where(has_msg) + LayerNorm ----------------
__global__ __launch_bounds__(256) void k_gru(
    const float* __restrict__ gi, const float* __restrict__ gh,
    const float* __restrict__ deg, const float* __restrict__ lng,
    const float* __restrict__ lnb, float* __restrict__ xf,
    unsigned short* __restrict__ xb) {
  __shared__ float ps[8], ps2[8];
  __shared__ float smean, sinv;
  int i = blockIdx.x, h = threadIdx.x;
  size_t b3 = (size_t)i * 768, b1 = (size_t)i * HD;
  float r = 1.f / (1.f + __expf(-(gi[b3 + h] + gh[b3 + h])));
  float z = 1.f / (1.f + __expf(-(gi[b3 + 256 + h] + gh[b3 + 256 + h])));
  float ng = tanhf(gi[b3 + 512 + h] + r * gh[b3 + 512 + h]);
  float xo = xf[b1 + h];
  float hn = (1.f - z) * ng + z * xo;
  float v = (deg[i] > 0.f) ? hn : xo;
  float s = v, s2 = v * v;
#pragma unroll
  for (int off = 16; off > 0; off >>= 1) { s += __shfl_down(s, off); s2 += __shfl_down(s2, off); }
  if ((threadIdx.x & 31) == 0) { ps[threadIdx.x >> 5] = s; ps2[threadIdx.x >> 5] = s2; }
  __syncthreads();
  if (threadIdx.x == 0) {
    float ts = 0.f, ts2 = 0.f;
    for (int j = 0; j < 8; ++j) { ts += ps[j]; ts2 += ps2[j]; }
    float m = ts / 256.f, var = ts2 / 256.f - m * m;
    smean = m; sinv = rsqrtf(var + 1e-5f);
  }
  __syncthreads();
  float y = (v - smean) * sinv * lng[h] + lnb[h];
  xf[b1 + h] = y;
  xb[b1 + h] = f2bf(y);
}

// ---------------- attention scores: tanh(x@W1+b1) . w2 + b2 ----------------
__global__ __launch_bounds__(256) void k_att(
    const unsigned short* __restrict__ xb, const unsigned short* __restrict__ W1t,
    const float* __restrict__ b1, const float* __restrict__ w2,
    const float* __restrict__ b2, float* __restrict__ scores) {
  __shared__ float sm[16][HD];
  __shared__ float red[16][16];
  int wave = threadIdx.x >> 5, lane = threadIdx.x & 31;
  int half = lane >> 4, l15 = lane & 15;
  int row0 = blockIdx.x * 16;
  const unsigned short* arow = xb + (size_t)(row0 + l15) * HD;
  const unsigned short* bcol0 = W1t + (size_t)(wave * 32 + l15) * HD;
  const unsigned short* bcol1 = bcol0 + (size_t)16 * HD;
  floatx8 acc0 = {}, acc1 = {};
  for (int kc = 0; kc < HD; kc += 32) {
    bf16x16 a = load_a(arow, kc, half);
    acc0 = wmma_bf(a, load_b(bcol0, kc, half), acc0);
    acc1 = wmma_bf(a, load_b(bcol1, kc, half), acc1);
  }
#pragma unroll
  for (int t = 0; t < 2; ++t) {
    int n = (wave * 2 + t) * 16 + l15;
    float bn = b1[n];
    floatx8 c = t ? acc1 : acc0;
#pragma unroll
    for (int r = 0; r < 8; ++r) sm[r + half * 8][n] = c[r] + bn;
  }
  __syncthreads();
  int rr = threadIdx.x >> 4, sub = threadIdx.x & 15;
  float s = 0.f;
#pragma unroll
  for (int j = 0; j < 16; ++j) { int n = sub * 16 + j; s += tanhf(sm[rr][n]) * w2[n]; }
  red[rr][sub] = s;
  __syncthreads();
  if (sub == 0) {
    float t = 0.f;
    for (int j = 0; j < 16; ++j) t += red[rr][j];
    scores[row0 + rr] = t + b2[0];
  }
}

// ---------------- per-graph softmax weights ----------------
__global__ __launch_bounds__(256) void k_softmax(
    const float* __restrict__ scores, const int* __restrict__ batch, float* __restrict__ w) {
  __shared__ float red[256];
  int g = blockIdx.x;
  float m = -1e30f;
  for (int i = threadIdx.x; i < NN; i += 256) if (batch[i] == g) m = fmaxf(m, scores[i]);
  red[threadIdx.x] = m; __syncthreads();
  for (int off = 128; off; off >>= 1) {
    if (threadIdx.x < off) red[threadIdx.x] = fmaxf(red[threadIdx.x], red[threadIdx.x + off]);
    __syncthreads();
  }
  float gm = red[0]; __syncthreads();
  float s = 0.f;
  for (int i = threadIdx.x; i < NN; i += 256) if (batch[i] == g) s += __expf(scores[i] - gm);
  red[threadIdx.x] = s; __syncthreads();
  for (int off = 128; off; off >>= 1) {
    if (threadIdx.x < off) red[threadIdx.x] += red[threadIdx.x + off];
    __syncthreads();
  }
  float denom = red[0];
  for (int i = threadIdx.x; i < NN; i += 256)
    if (batch[i] == g) w[i] = __expf(scores[i] - gm) / denom;
}

__global__ void k_pool(const float* __restrict__ xf, const float* __restrict__ w,
                       const int* __restrict__ batch, float* __restrict__ pooled) {
  int i = blockIdx.x, h = threadIdx.x;
  atomicAdd(&pooled[(size_t)batch[i] * HD + h], xf[(size_t)i * HD + h] * w[i]);
}

// ---------------- projection 16x512 (K=256) + LayerNorm ----------------
__global__ __launch_bounds__(256) void k_proj(
    const unsigned short* __restrict__ pb, const unsigned short* __restrict__ Wt,
    const float* __restrict__ bias, const float* __restrict__ g,
    const float* __restrict__ beta, float* __restrict__ out) {
  __shared__ float sm[16][OUTF];
  __shared__ float red[2][16][16];
  __shared__ float mn[16], iv[16];
  int wave = threadIdx.x >> 5, lane = threadIdx.x & 31;
  int half = lane >> 4, l15 = lane & 15;
  const unsigned short* arow = pb + (size_t)l15 * HD;
  const unsigned short* bc = Wt + (size_t)(wave * 64 + l15) * HD;
  floatx8 acc[4] = {{}, {}, {}, {}};
  for (int kc = 0; kc < HD; kc += 32) {
    bf16x16 a = load_a(arow, kc, half);
#pragma unroll
    for (int j = 0; j < 4; ++j)
      acc[j] = wmma_bf(a, load_b(bc + (size_t)j * 16 * HD, kc, half), acc[j]);
  }
#pragma unroll
  for (int j = 0; j < 4; ++j) {
    int n = wave * 64 + j * 16 + l15;
    float bn = bias[n];
#pragma unroll
    for (int r = 0; r < 8; ++r) sm[r + half * 8][n] = acc[j][r] + bn;
  }
  __syncthreads();
  int rr = threadIdx.x >> 4, sub = threadIdx.x & 15;
  float s = 0.f, s2 = 0.f;
#pragma unroll
  for (int j = 0; j < 32; ++j) { float v = sm[rr][sub * 32 + j]; s += v; s2 += v * v; }
  red[0][rr][sub] = s; red[1][rr][sub] = s2;
  __syncthreads();
  if (sub == 0) {
    float ts = 0.f, ts2 = 0.f;
    for (int j = 0; j < 16; ++j) { ts += red[0][rr][j]; ts2 += red[1][rr][j]; }
    float m = ts / 512.f, var = ts2 / 512.f - m * m;
    mn[rr] = m; iv[rr] = rsqrtf(var + 1e-5f);
  }
  __syncthreads();
#pragma unroll
  for (int j = 0; j < 32; ++j) {
    int n = sub * 32 + j;
    out[(size_t)rr * OUTF + n] = (sm[rr][n] - mn[rr]) * iv[rr] * g[n] + beta[n];
  }
}

// ---------------- host ----------------
extern "C" void kernel_launch(void* const* d_in, const int* in_sizes, int n_in,
                              void* d_out, int out_size, void* d_ws, size_t ws_size,
                              hipStream_t stream) {
  const float* nf        = (const float*)d_in[0];
  const int*   ei        = (const int*)d_in[1];
  const float* ef        = (const float*)d_in[2];
  const int*   batch     = (const int*)d_in[3];
  const float* node_w    = (const float*)d_in[4];
  const float* node_b    = (const float*)d_in[5];
  const float* node_g    = (const float*)d_in[6];
  const float* node_beta = (const float*)d_in[7];
  const float* edge_w    = (const float*)d_in[8];
  const float* edge_b    = (const float*)d_in[9];
  const float* edge_g    = (const float*)d_in[10];
  const float* edge_beta = (const float*)d_in[11];
  const float* mp_w1     = (const float*)d_in[12];
  const float* mp_b1     = (const float*)d_in[13];
  const float* mp_w2     = (const float*)d_in[14];
  const float* mp_b2     = (const float*)d_in[15];
  const float* gru_wih   = (const float*)d_in[16];
  const float* gru_whh   = (const float*)d_in[17];
  const float* gru_bih   = (const float*)d_in[18];
  const float* gru_bhh   = (const float*)d_in[19];
  const float* mp_ln_g   = (const float*)d_in[20];
  const float* mp_ln_b   = (const float*)d_in[21];
  const float* att_w1    = (const float*)d_in[22];
  const float* att_b1    = (const float*)d_in[23];
  const float* att_w2    = (const float*)d_in[24];
  const float* att_b2    = (const float*)d_in[25];
  const float* proj_w    = (const float*)d_in[26];
  const float* proj_b    = (const float*)d_in[27];
  const float* proj_g    = (const float*)d_in[28];
  const float* proj_beta = (const float*)d_in[29];

  const int* src  = ei;
  const int* dstI = ei + NE;

  char* base = (char*)d_ws;
  size_t off = 0;
  auto alloc = [&](size_t bytes) -> void* {
    void* p = base + off;
    off += (bytes + 255) & ~(size_t)255;
    return p;
  };
  unsigned short* xb     = (unsigned short*)alloc((size_t)NN * HD * 2);
  float*          xf     = (float*)         alloc((size_t)NN * HD * 4);
  unsigned short* ebuf   = (unsigned short*)alloc((size_t)NE * HD * 2);
  float*          agg    = (float*)         alloc((size_t)NN * HD * 4);
  unsigned short* aggb   = (unsigned short*)alloc((size_t)NN * HD * 2);
  float*          gi     = (float*)         alloc((size_t)NN * 768 * 4);
  float*          gh     = (float*)         alloc((size_t)NN * 768 * 4);
  float*          deg    = (float*)         alloc((size_t)NN * 4);
  float*          scores = (float*)         alloc((size_t)NN * 4);
  float*          attw   = (float*)         alloc((size_t)NN * 4);
  float*          pooled = (float*)         alloc((size_t)NG * HD * 4);
  unsigned short* pooledb= (unsigned short*)alloc((size_t)NG * HD * 2);
  unsigned short* nfb    = (unsigned short*)alloc((size_t)NN * 64 * 2);
  unsigned short* efb    = (unsigned short*)alloc((size_t)NE * 32 * 2);
  unsigned short* nwt    = (unsigned short*)alloc((size_t)256 * 64 * 2);
  unsigned short* ewt    = (unsigned short*)alloc((size_t)256 * 32 * 2);
  unsigned short* w1t    = (unsigned short*)alloc((size_t)3 * 256 * 768 * 2);
  unsigned short* w2t    = (unsigned short*)alloc((size_t)3 * 256 * 256 * 2);
  unsigned short* wiht   = (unsigned short*)alloc((size_t)3 * 768 * 256 * 2);
  unsigned short* whht   = (unsigned short*)alloc((size_t)3 * 768 * 256 * 2);
  unsigned short* att1t  = (unsigned short*)alloc((size_t)256 * 256 * 2);
  unsigned short* projt  = (unsigned short*)alloc((size_t)512 * 256 * 2);

  auto cdiv = [](int a, int b) { return (a + b - 1) / b; };

  // --- prep: bf16 conversions + weight transposes ---
  k_f2bf<<<cdiv(NN * 64, 256), 256, 0, stream>>>(nf, nfb, NN * 64);
  k_f2bf<<<cdiv(NE * 32, 256), 256, 0, stream>>>(ef, efb, NE * 32);
  k_tconv<<<cdiv(64 * 256, 256), 256, 0, stream>>>(node_w, nwt, 64, 256);
  k_tconv<<<cdiv(32 * 256, 256), 256, 0, stream>>>(edge_w, ewt, 32, 256);
  for (int l = 0; l < 3; ++l) {
    k_tconv<<<cdiv(768 * 256, 256), 256, 0, stream>>>(mp_w1 + (size_t)l * 768 * 256, w1t + (size_t)l * 256 * 768, 768, 256);
    k_tconv<<<cdiv(256 * 256, 256), 256, 0, stream>>>(mp_w2 + (size_t)l * 256 * 256, w2t + (size_t)l * 256 * 256, 256, 256);
    k_tconv<<<cdiv(256 * 768, 256), 256, 0, stream>>>(gru_wih + (size_t)l * 256 * 768, wiht + (size_t)l * 768 * 256, 256, 768);
    k_tconv<<<cdiv(256 * 768, 256), 256, 0, stream>>>(gru_whh + (size_t)l * 256 * 768, whht + (size_t)l * 768 * 256, 256, 768);
  }
  k_tconv<<<cdiv(256 * 256, 256), 256, 0, stream>>>(att_w1, att1t, 256, 256);
  k_tconv<<<cdiv(256 * 512, 256), 256, 0, stream>>>(proj_w, projt, 256, 512);

  // --- degree ---
  k_zero<<<cdiv(NN, 256), 256, 0, stream>>>(deg, NN);
  k_deg<<<cdiv(NE, 256), 256, 0, stream>>>(dstI, deg, NE);

  // --- encoders ---
  k_encode<<<NN / 16, 256, 0, stream>>>(nfb, nwt, node_b, node_g, node_beta, xb, xf, 64);
  k_encode<<<NE / 16, 256, 0, stream>>>(efb, ewt, edge_b, edge_g, edge_beta, ebuf, nullptr, 32);

  // --- message passing layers ---
  for (int l = 0; l < 3; ++l) {
    k_zero<<<cdiv(NN * HD, 256), 256, 0, stream>>>(agg, NN * HD);
    k_message<<<NE / 16, 256, 0, stream>>>(xb, ebuf, src, dstI,
        w1t + (size_t)l * 256 * 768, mp_b1 + (size_t)l * 256,
        w2t + (size_t)l * 256 * 256, mp_b2 + (size_t)l * 256, agg);
    k_aggfix<<<NN, 256, 0, stream>>>(agg, deg, aggb);
    {
      dim3 g1(NN / 16, 768 / 64);
      k_gemm<<<g1, 32, 0, stream>>>(aggb, wiht + (size_t)l * 768 * 256, gru_bih + (size_t)l * 768, gi, 256, 768);
      k_gemm<<<g1, 32, 0, stream>>>(xb,   whht + (size_t)l * 768 * 256, gru_bhh + (size_t)l * 768, gh, 256, 768);
    }
    k_gru<<<NN, 256, 0, stream>>>(gi, gh, deg, mp_ln_g + (size_t)l * 256, mp_ln_b + (size_t)l * 256, xf, xb);
  }

  // --- attention pooling ---
  k_att<<<NN / 16, 256, 0, stream>>>(xb, att1t, att_b1, att_w2, att_b2, scores);
  k_softmax<<<NG, 256, 0, stream>>>(scores, batch, attw);
  k_zero<<<cdiv(NG * HD, 256), 256, 0, stream>>>(pooled, NG * HD);
  k_pool<<<NN, 256, 0, stream>>>(xf, attw, batch, pooled);
  k_f2bf<<<cdiv(NG * HD, 256), 256, 0, stream>>>(pooled, pooledb, NG * HD);

  // --- projection + final LN ---
  k_proj<<<1, 256, 0, stream>>>(pooledb, projt, proj_b, proj_g, proj_beta, (float*)d_out);
}
